// FlowSeekEngine_40905268527529
// MI455X (gfx1250) — compile-verified, matrix operand order checked by reading, not run on hardware
//
#include <hip/hip_runtime.h>
#include <hip/hip_bf16.h>
#include <math.h>

typedef __attribute__((ext_vector_type(16))) _Float16 v16h;
typedef __attribute__((ext_vector_type(8)))  float    v8f;

#define CCH 256   // channels
#define RAD 4     // radius
#define NDISP 81  // (2*RAD+1)^2

// ---------------------------------------------------------------------------
// Kernel 1: avg-pool (s x s) from the original fp32 map, L2-normalize over
// channels, emit f16 in pixel-major channel-contiguous layout:
//   dst[((b*Hl + h)*Wl + w)*256 + c]
// One block (256 threads == 256 channels) per output pixel.
// ---------------------------------------------------------------------------
__global__ __launch_bounds__(256)
void fs_pool_norm_kernel(const float* __restrict__ src, _Float16* __restrict__ dst,
                         int H, int W, int s, int Hl, int Wl)
{
    const int pix = blockIdx.x;              // b*Hl*Wl + h*Wl + w
    const int c   = threadIdx.x;             // channel
    const int w   = pix % Wl;
    const int t2  = pix / Wl;
    const int h   = t2 % Hl;
    const int b   = t2 / Hl;

    const float* p = src + (((size_t)b * CCH + c) * H + (size_t)h * s) * W + (size_t)w * s;
    float acc = 0.0f;
    for (int i = 0; i < s; ++i)
        for (int j = 0; j < s; ++j)
            acc += p[(size_t)i * W + j];
    acc *= 1.0f / (float)(s * s);

    // block-wide sum of squares: wave32 shuffle reduce, then LDS across 8 waves
    float sq = acc * acc;
    #pragma unroll
    for (int off = 16; off > 0; off >>= 1)
        sq += __shfl_xor(sq, off, 32);

    __shared__ float red[8];
    const int wid = threadIdx.x >> 5;
    if ((threadIdx.x & 31) == 0) red[wid] = sq;
    __syncthreads();
    float norm2 = 0.0f;
    #pragma unroll
    for (int i = 0; i < 8; ++i) norm2 += red[i];

    const float inv = 1.0f / fmaxf(sqrtf(norm2), 1e-12f);
    dst[(size_t)pix * CCH + c] = (_Float16)(acc * inv);
}

// ---------------------------------------------------------------------------
// Kernel 2: per-pixel depth bias = depth_w * exp(-|avgpool(d1)-avgpool(d2)|)
// ---------------------------------------------------------------------------
__global__ __launch_bounds__(256)
void fs_depth_bias_kernel(const float* __restrict__ d1, const float* __restrict__ d2,
                          const float* __restrict__ dwp, float* __restrict__ bias,
                          int H, int W, int s, int Hl, int Wl, int total)
{
    const int idx = blockIdx.x * 256 + threadIdx.x;
    if (idx >= total) return;
    const int w  = idx % Wl;
    const int t  = idx / Wl;
    const int h  = t % Hl;
    const int b  = t / Hl;

    const float* p1 = d1 + ((size_t)b * H + (size_t)h * s) * W + (size_t)w * s;
    const float* p2 = d2 + ((size_t)b * H + (size_t)h * s) * W + (size_t)w * s;
    float a1 = 0.0f, a2 = 0.0f;
    for (int i = 0; i < s; ++i)
        for (int j = 0; j < s; ++j) {
            a1 += p1[(size_t)i * W + j];
            a2 += p2[(size_t)i * W + j];
        }
    const float invn = 1.0f / (float)(s * s);
    a1 *= invn; a2 *= invn;
    bias[idx] = dwp[0] * expf(-fabsf(a1 - a2));
}

// ---------------------------------------------------------------------------
// Kernel 3: WMMA local correlation.
// One wave per 4x4 f1-pixel tile. A = 16 f1 pixels x 256 ch (f16).
// B = 12x12 f2 neighborhood -> 9 column tiles of 16 pixels x 256 ch.
// Chain 8x v_wmma_f32_16x16x32_f16 over K per N-tile. Interior tiles take a
// select-free fast path (wave-uniform __all in-bounds test); edge tiles zero
// out-of-image columns to match the reference's zero padding. Scatter the
// valid (pixel, displacement) pairs with the hoisted depth bias added.
// ---------------------------------------------------------------------------
__global__ __launch_bounds__(128)
void fs_corr_wmma_kernel(const _Float16* __restrict__ f1n,
                         const _Float16* __restrict__ f2n,
                         const float* __restrict__ bias,
                         float* __restrict__ out,       // level base
                         int Hl, int Wl, int numTiles)
{
    const int wave = threadIdx.x >> 5;
    const int lane = threadIdx.x & 31;
    const int tile = blockIdx.x * 4 + wave;
    if (tile >= numTiles) return;                       // uniform per wave

    const int tilesW = Wl >> 2;
    const int tilesH = Hl >> 2;
    const int tw = tile % tilesW;
    const int r1 = tile / tilesW;
    const int th = r1 % tilesH;
    const int b  = r1 / tilesH;
    const int h0 = th * 4, w0 = tw * 4;

    const int mp = lane & 15;    // A: row pixel | B: column pixel | D: N index
    const int kg = lane >> 4;    // K half-group (0: K0..15, 1: K16..31)

    // ---- preload A: f1 tile, pixel mp -> (h0 + mp/4, w0 + mp%4) ----
    const int ah = h0 + (mp >> 2);
    const int aw = w0 + (mp & 3);
    const _Float16* aptr =
        f1n + (((size_t)b * Hl + ah) * Wl + aw) * CCH + kg * 16;
    v16h A[8];
    #pragma unroll
    for (int k = 0; k < 8; ++k)
        A[k] = *(const v16h*)(aptr + k * 32);

    // ---- hoist per-lane depth bias for the 8 output pixels it scatters ----
    float biasv[8];
    #pragma unroll
    for (int r = 0; r < 8; ++r) {
        const int m  = r + kg * 8;
        const int h1 = h0 + (m >> 2);
        const int w1 = w0 + (m & 3);
        biasv[r] = bias[((size_t)b * Hl + h1) * Wl + w1];
    }

    const v16h zeroB = {};

    #pragma unroll 1
    for (int t = 0; t < 9; ++t) {
        // B column pixel: n in 0..143 -> 12x12 patch at (h0-4, w0-4)
        const int n  = t * 16 + mp;
        const int h2 = h0 - RAD + n / 12;
        const int w2 = w0 - RAD + n % 12;
        const bool inb = (h2 >= 0) & (h2 < Hl) & (w2 >= 0) & (w2 < Wl);
        const int h2c = min(max(h2, 0), Hl - 1);
        const int w2c = min(max(w2, 0), Wl - 1);
        const _Float16* bptr =
            f2n + (((size_t)b * Hl + h2c) * Wl + w2c) * CCH + kg * 16;

        // prime the cache pipe for the next N-tile (global_prefetch_b8)
        if (t < 8) {
            const int nn  = (t + 1) * 16 + mp;
            const int nh  = min(max(h0 - RAD + nn / 12, 0), Hl - 1);
            const int nw  = min(max(w0 - RAD + nn % 12, 0), Wl - 1);
            __builtin_prefetch(
                f2n + (((size_t)b * Hl + nh) * Wl + nw) * CCH + kg * 16, 0, 0);
        }

        v8f acc = {};
        if (__all((int)inb)) {
            // fast path: whole 16-pixel column tile is in-image, no selects
            #pragma unroll
            for (int k = 0; k < 8; ++k) {
                const v16h Bv = *(const v16h*)(bptr + k * 32);
                acc = __builtin_amdgcn_wmma_f32_16x16x32_f16(
                          false, A[k], false, Bv, (short)0, acc, false, false);
            }
        } else {
            // edge path: clamped-address loads, zero-select out-of-image lanes
            #pragma unroll
            for (int k = 0; k < 8; ++k) {
                v16h Bv = *(const v16h*)(bptr + k * 32);
                if (!inb) Bv = zeroB;                   // zero-padding semantics
                acc = __builtin_amdgcn_wmma_f32_16x16x32_f16(
                          false, A[k], false, Bv, (short)0, acc, false, false);
            }
        }

        // ---- scatter: D VGPR r holds (M = r + 8*kg, N = mp) ----
        #pragma unroll
        for (int r = 0; r < 8; ++r) {
            const int m  = r + kg * 8;                  // f1 pixel in tile
            const int h1 = h0 + (m >> 2);
            const int w1 = w0 + (m & 3);
            const int dy = (n / 12) - RAD - (m >> 2);
            const int dx = (n % 12) - RAD - (m & 3);
            if (dy >= -RAD && dy <= RAD && dx >= -RAD && dx <= RAD) {
                const int d = (dy + RAD) * 9 + (dx + RAD);
                const size_t oidx =
                    (((size_t)b * NDISP + d) * Hl + h1) * Wl + w1;
                out[oidx] = acc[r] + biasv[r];
            }
        }
    }
}

// ---------------------------------------------------------------------------
extern "C" void kernel_launch(void* const* d_in, const int* in_sizes, int n_in,
                              void* d_out, int out_size, void* d_ws, size_t ws_size,
                              hipStream_t stream)
{
    const float* fmap1   = (const float*)d_in[0];
    const float* fmap2   = (const float*)d_in[1];
    const float* depth1  = (const float*)d_in[2];
    const float* depth2  = (const float*)d_in[3];
    const float* depth_w = (const float*)d_in[4];
    float* out = (float*)d_out;

    const int B = 4, H = 160, W = 160;
    const int Hl[4] = {160, 80, 40, 20};
    const int Wl[4] = {160, 80, 40, 20};

    // workspace layout: f16 f1 pyramid | f16 f2 pyramid | fp32 bias pyramid
    size_t pixCnt[4], halfCnt[4];
    for (int l = 0; l < 4; ++l) {
        pixCnt[l]  = (size_t)B * Hl[l] * Wl[l];
        halfCnt[l] = pixCnt[l] * CCH;
    }
    size_t f1off[4], f2off[4], cur = 0;
    for (int l = 0; l < 4; ++l) { f1off[l] = cur; cur += halfCnt[l]; }
    for (int l = 0; l < 4; ++l) { f2off[l] = cur; cur += halfCnt[l]; }
    _Float16* wsH      = (_Float16*)d_ws;
    float*    biasBase = (float*)(wsH + cur);       // cur is even -> 4B aligned
    size_t boff[4], bc = 0;
    for (int l = 0; l < 4; ++l) { boff[l] = bc; bc += pixCnt[l]; }

    const size_t outOff[4] = {0, 8294400, 10368000, 10886400};

    for (int l = 0; l < 4; ++l) {
        const int s = 1 << l;
        const int total = (int)pixCnt[l];

        fs_pool_norm_kernel<<<total, 256, 0, stream>>>(
            fmap1, wsH + f1off[l], H, W, s, Hl[l], Wl[l]);
        fs_pool_norm_kernel<<<total, 256, 0, stream>>>(
            fmap2, wsH + f2off[l], H, W, s, Hl[l], Wl[l]);

        fs_depth_bias_kernel<<<(total + 255) / 256, 256, 0, stream>>>(
            depth1, depth2, depth_w, biasBase + boff[l],
            H, W, s, Hl[l], Wl[l], total);

        const int numTiles = B * (Hl[l] / 4) * (Wl[l] / 4);
        fs_corr_wmma_kernel<<<(numTiles + 3) / 4, 128, 0, stream>>>(
            wsH + f1off[l], wsH + f2off[l], biasBase + boff[l],
            out + outOff[l], Hl[l], Wl[l], numTiles);
    }
}
